// Copy_Decoder_42640435314947
// MI455X (gfx1250) — matrix operand order repaired and checked by reading, not compiled, not run on hardware
//
#include <hip/hip_runtime.h>
#include <math.h>

// ---------------------------------------------------------------------------
// Copy-decoder step for MI455X (gfx1250, wave32).
//  * All batch-1 GEMVs -> bandwidth-bound wave-per-row kernels (~270MB total,
//    ~12us at 23.3 TB/s; W_gen GEMV at 206MB dominates).
//  * The 2048x1024x1024 GEMM inside score_c (tanh between matmuls blocks
//    reassociation) runs on v_wmma_f32_16x16x32_bf16, one wave per 16x16
//    tile, fused tanh/dot-h0 epilogue so the 8MB intermediate never exists.
// ---------------------------------------------------------------------------

constexpr int kV  = 50257;
constexpr int kH  = 1024;
constexpr int kL  = 2048;
constexpr int kML = 2048;

typedef __attribute__((ext_vector_type(16))) __bf16 v16bf;
typedef __attribute__((ext_vector_type(8)))  float  v8f;

// Workspace layout (float offsets)
constexpr size_t OFF_EMB0  = 0;                    // 1024
constexpr size_t OFF_CAT   = 1024;                 // 2048  [emb0 ; h0]
constexpr size_t OFF_ATTN  = 3072;                 // 2048  attn scores -> weights
constexpr size_t OFF_JOIN  = 5120;                 // 2048  [emb0 ; attn_applied]
constexpr size_t OFF_RNNIN = 7168;                 // 1024
constexpr size_t OFF_GI    = 8192;                 // 3072
constexpr size_t OFF_GH    = 11264;                // 3072
constexpr size_t OFF_HNEW  = 14336;                // 1024
constexpr size_t OFF_WVEC  = 15360;                // 2048  prob_c * matched
constexpr size_t OFF_RED   = 17408;                // 4     [max, invZ]
constexpr size_t OFF_SG    = 17920;                // 50257 score_g
constexpr size_t OFF_SC    = 68608;                // 2048  score_c (atomically accumulated)
constexpr size_t OFF_ABF   = 70656;                // enc as bf16: 2048*1024 u16 (=1048576 f)
constexpr size_t OFF_BBF   = 70656 + 1048576;      // W_copy as bf16: 1024*1024 u16
// total ~1.64M floats (~6.3 MB) of scratch

__device__ __forceinline__ unsigned short f2bf(float f) {
  unsigned u = __float_as_uint(f);
  unsigned r = u + 0x7FFFu + ((u >> 16) & 1u);     // round-to-nearest-even
  return (unsigned short)(r >> 16);
}

// ---- init: gather embedding row, build concat inputs, zero accumulators ----
__global__ void k_init(const int* __restrict__ din, const float* __restrict__ hid,
                       const float* __restrict__ emb, float* __restrict__ ws,
                       float* __restrict__ out_fw) {
  int t = blockIdx.x * blockDim.x + threadIdx.x;   // 0..2047
  int d = din[0];
  if (t < kH) {
    float e = emb[(size_t)d * kH + t];
    ws[OFF_EMB0 + t]      = e;
    ws[OFF_CAT  + t]      = e;
    ws[OFF_CAT  + kH + t] = hid[t];
    ws[OFF_JOIN + t]      = e;
    ws[OFF_JOIN + kH + t] = 0.0f;                  // attn_applied accumulator
    out_fw[t]             = 0.0f;                  // final_weights accumulator
  }
  if (t < kL) ws[OFF_SC + t] = 0.0f;               // score_c accumulator
}

// ---- generic GEMV: y = act(W @ x + b), one wave32 per row ------------------
__global__ void k_gemv(const float* __restrict__ W, const float* __restrict__ x,
                       const float* __restrict__ b, float* __restrict__ y,
                       int R, int K, int act) {
  int wave = threadIdx.x >> 5;
  int lane = threadIdx.x & 31;
  int row  = blockIdx.x * (blockDim.x >> 5) + wave;
  if (row >= R) return;
  const float4* w4 = (const float4*)(W + (size_t)row * K);
  const float4* x4 = (const float4*)x;
  int K4 = K >> 2;
  float s = 0.0f;
  for (int i = lane; i < K4; i += 32) {
    float4 wv = w4[i]; float4 xv = x4[i];
    s += wv.x * xv.x + wv.y * xv.y + wv.z * xv.z + wv.w * xv.w;
  }
  #pragma unroll
  for (int m = 16; m >= 1; m >>= 1) s += __shfl_xor(s, m, 32);
  if (lane == 0) {
    s += b[row];
    if (act == 1) s = s > 0.0f ? s : 0.0f;
    y[row] = s;
  }
}

// ---- single-block in-place softmax over N elements -------------------------
__global__ void k_softmax(float* __restrict__ s, int N) {
  __shared__ float red[256];
  int t = threadIdx.x;
  float mx = -INFINITY;
  for (int i = t; i < N; i += 256) mx = fmaxf(mx, s[i]);
  red[t] = mx; __syncthreads();
  for (int o = 128; o > 0; o >>= 1) { if (t < o) red[t] = fmaxf(red[t], red[t + o]); __syncthreads(); }
  mx = red[0]; __syncthreads();
  float sum = 0.0f;
  for (int i = t; i < N; i += 256) { float e = __expf(s[i] - mx); s[i] = e; sum += e; }
  red[t] = sum; __syncthreads();
  for (int o = 128; o > 0; o >>= 1) { if (t < o) red[t] += red[t + o]; __syncthreads(); }
  float inv = 1.0f / red[0];
  for (int i = t; i < N; i += 256) s[i] *= inv;
}

// ---- attn_applied[h] = sum_m aw[m] * enc[m,h]  (partials via atomics) ------
__global__ void k_attn_apply(const float* __restrict__ aw, const float* __restrict__ enc,
                             float* __restrict__ joined) {
  int h  = blockIdx.x * 256 + threadIdx.x;         // gridDim.x = 4 -> h in [0,1024)
  int m0 = blockIdx.y * 256;                       // gridDim.y = 8
  float s = 0.0f;
  for (int m = m0; m < m0 + 256; ++m) s += aw[m] * enc[(size_t)m * kH + h];
  atomicAdd(&joined[kH + h], s);
}

// ---- GRU gates -------------------------------------------------------------
__global__ void k_gru(const float* __restrict__ gi, const float* __restrict__ gh,
                      const float* __restrict__ h0, float* __restrict__ hnew,
                      float* __restrict__ out_hidden) {
  int t = blockIdx.x * blockDim.x + threadIdx.x;
  if (t >= kH) return;
  float r = 1.0f / (1.0f + __expf(-(gi[t] + gh[t])));
  float z = 1.0f / (1.0f + __expf(-(gi[kH + t] + gh[kH + t])));
  float n = tanhf(gi[2 * kH + t] + r * gh[2 * kH + t]);
  float hn = (1.0f - z) * n + z * h0[t];
  hnew[t] = hn;
  out_hidden[t] = hn;
}

// ---- f32 -> bf16 staging ---------------------------------------------------
__global__ void k_cvt(const float* __restrict__ src, unsigned short* __restrict__ dst, int n) {
  int i = blockIdx.x * blockDim.x + threadIdx.x;
  if (i < n) dst[i] = f2bf(src[i]);
}

// ---- score_c[i] = sum_j tanh((enc @ W_copy^T)[i,j] + b_copy[j]) * h0[j] ----
// One wave per 16x16 output tile; K loop of 32 x v_wmma_f32_16x16x32_bf16.
// A operand = enc tile (row-major bf16), B operand = W_copy rows (col j of B).
__global__ void k_copy_score(const unsigned short* __restrict__ A,   // 2048 x 1024 bf16
                             const unsigned short* __restrict__ B,   // 1024 x 1024 bf16 (W_copy)
                             const float* __restrict__ bias,
                             const float* __restrict__ h0,
                             float* __restrict__ score_c) {
  int lane = threadIdx.x;                  // 0..31, wave fully active (EXEC all 1s)
  int tj   = blockIdx.x;                   // col tile 0..63
  int ti   = blockIdx.y;                   // row tile 0..127
  int half = lane >> 4;                    // 0 / 1
  int l15  = lane & 15;

  const unsigned short* arow = A + (size_t)(ti * 16 + l15) * kH;  // M = l15
  const unsigned short* brow = B + (size_t)(tj * 16 + l15) * kH;  // N = l15

  v8f acc = {};
  for (int k0 = 0; k0 < kH; k0 += 32) {
    union { v16bf v; uint4 u[2]; } a, b;
    // A 16x32: elems 0-7 = K[k0+half*8 .. +7], elems 8-15 = K[k0+16+half*8 .. +7]
    a.u[0] = *(const uint4*)(arow + k0 + half * 8);
    a.u[1] = *(const uint4*)(arow + k0 + 16 + half * 8);
    // B 32x16: elems 0-7 = K[k0+half*8 .. +7], elems 8-15 = K[k0+16+half*8 .. +7] for column l15
    b.u[0] = *(const uint4*)(brow + k0 + half * 8);
    b.u[1] = *(const uint4*)(brow + k0 + 16 + half * 8);
    acc = __builtin_amdgcn_wmma_f32_16x16x32_bf16(
        /*neg_a=*/false, a.v, /*neg_b=*/false, b.v,
        /*c_mod=*/(short)0, acc, /*reuse_a=*/false, /*reuse_b=*/false);
  }

  // C layout: acc[r] holds row (r + 8*half), col l15 of the tile.
  float bb = bias[tj * 16 + l15];
  float hh = h0[tj * 16 + l15];
  #pragma unroll
  for (int r = 0; r < 8; ++r) {
    float v = tanhf(acc[r] + bb) * hh;
    v += __shfl_xor(v, 1, 32);
    v += __shfl_xor(v, 2, 32);
    v += __shfl_xor(v, 4, 32);
    v += __shfl_xor(v, 8, 32);             // reduce 16 cols within each half-wave
    if (l15 == 0) atomicAdd(&score_c[ti * 16 + r + 8 * half], v);
  }
}

// ---- joint softmax stats over [score_g ; score_c] --------------------------
__global__ void k_stats(const float* __restrict__ sg, const float* __restrict__ sc,
                        float* __restrict__ red) {
  __shared__ float sh[256];
  int t = threadIdx.x;
  float mx = -INFINITY;
  for (int i = t; i < kV; i += 256) mx = fmaxf(mx, sg[i]);
  for (int i = t; i < kL; i += 256) mx = fmaxf(mx, sc[i]);
  sh[t] = mx; __syncthreads();
  for (int o = 128; o > 0; o >>= 1) { if (t < o) sh[t] = fmaxf(sh[t], sh[t + o]); __syncthreads(); }
  mx = sh[0]; __syncthreads();
  float sum = 0.0f;
  for (int i = t; i < kV; i += 256) sum += __expf(sg[i] - mx);
  for (int i = t; i < kL; i += 256) sum += __expf(sc[i] - mx);
  sh[t] = sum; __syncthreads();
  for (int o = 128; o > 0; o >>= 1) { if (t < o) sh[t] += sh[t + o]; __syncthreads(); }
  if (t == 0) { red[0] = mx; red[1] = 1.0f / sh[0]; }
}

// ---- prob_g into d_out -----------------------------------------------------
__global__ void k_probg(const float* __restrict__ sg, const float* __restrict__ red,
                        float* __restrict__ out) {
  int v = blockIdx.x * 256 + threadIdx.x;
  if (v < kV) out[v] = __expf(sg[v] - red[0]) * red[1];
}

// ---- prob_c scatter-add + matched weight vector ----------------------------
__global__ void k_scatter(const float* __restrict__ sc, const float* __restrict__ red,
                          const int* __restrict__ idx, const int* __restrict__ din,
                          float* __restrict__ out_final, float* __restrict__ wvec) {
  int l = blockIdx.x * 256 + threadIdx.x;
  if (l >= kL) return;
  float p = __expf(sc[l] - red[0]) * red[1];
  atomicAdd(&out_final[idx[l]], p);
  wvec[l] = (idx[l] == din[0]) ? p : 0.0f;
}

// ---- final_weights[h] = sum_l wvec[l] * enc[l,h] ---------------------------
__global__ void k_fw(const float* __restrict__ wvec, const float* __restrict__ enc,
                     float* __restrict__ out_fw) {
  int h  = blockIdx.x * 256 + threadIdx.x;         // gridDim.x = 4
  int m0 = blockIdx.y * 256;                       // gridDim.y = 8
  float s = 0.0f;
  for (int m = m0; m < m0 + 256; ++m) s += wvec[m] * enc[(size_t)m * kH + h];
  atomicAdd(&out_fw[h], s);
}

extern "C" void kernel_launch(void* const* d_in, const int* in_sizes, int n_in,
                              void* d_out, int out_size, void* d_ws, size_t ws_size,
                              hipStream_t stream) {
  const int*   din      = (const int*)  d_in[0];
  const float* hid      = (const float*)d_in[1];   // h0, length H
  const float* enc      = (const float*)d_in[2];   // ML x H
  const int*   idx      = (const int*)  d_in[3];   // L
  const float* emb      = (const float*)d_in[4];   // V x H
  const float* W_attn_w = (const float*)d_in[5];
  const float* b_attn_w = (const float*)d_in[6];
  const float* W_attn_u = (const float*)d_in[7];
  const float* b_attn_u = (const float*)d_in[8];
  const float* W_ih     = (const float*)d_in[9];
  const float* b_ih     = (const float*)d_in[10];
  const float* W_hh     = (const float*)d_in[11];
  const float* b_hh     = (const float*)d_in[12];
  const float* W_copy   = (const float*)d_in[13];
  const float* b_copy   = (const float*)d_in[14];
  const float* W_gen    = (const float*)d_in[15];
  const float* b_gen    = (const float*)d_in[16];

  float* out = (float*)d_out;          // [0,V)=final_output  [V,V+H)=hidden  [V+H,V+2H)=weights
  float* ws  = (float*)d_ws;
  unsigned short* Abf = (unsigned short*)(ws + OFF_ABF);
  unsigned short* Bbf = (unsigned short*)(ws + OFF_BBF);

  k_init<<<8, 256, 0, stream>>>(din, hid, emb, ws, out + kV + kH);

  // attn scores (2048 rows, K=2048), softmax, context
  k_gemv<<<256, 256, 0, stream>>>(W_attn_w, ws + OFF_CAT, b_attn_w, ws + OFF_ATTN, kML, 2 * kH, 0);
  k_softmax<<<1, 256, 0, stream>>>(ws + OFF_ATTN, kML);
  k_attn_apply<<<dim3(4, 8), 256, 0, stream>>>(ws + OFF_ATTN, enc, ws + OFF_JOIN);

  // rnn_in = relu(W_attn_u @ joined + b)
  k_gemv<<<128, 256, 0, stream>>>(W_attn_u, ws + OFF_JOIN, b_attn_u, ws + OFF_RNNIN, kH, 2 * kH, 1);

  // GRU
  k_gemv<<<384, 256, 0, stream>>>(W_ih, ws + OFF_RNNIN, b_ih, ws + OFF_GI, 3 * kH, kH, 0);
  k_gemv<<<384, 256, 0, stream>>>(W_hh, hid,            b_hh, ws + OFF_GH, 3 * kH, kH, 0);
  k_gru<<<4, 256, 0, stream>>>(ws + OFF_GI, ws + OFF_GH, hid, ws + OFF_HNEW, out + kV);

  // score_g = W_gen @ h_new + b_gen   (206 MB, the bandwidth pole)
  k_gemv<<<(kV + 7) / 8, 256, 0, stream>>>(W_gen, ws + OFF_HNEW, b_gen, ws + OFF_SG, kV, kH, 0);

  // score_c via bf16 WMMA GEMM with fused tanh/dot epilogue
  k_cvt<<<(kL * kH + 255) / 256, 256, 0, stream>>>(enc,    Abf, kL * kH);
  k_cvt<<<(kH * kH + 255) / 256, 256, 0, stream>>>(W_copy, Bbf, kH * kH);
  k_copy_score<<<dim3(kH / 16, kL / 16), 32, 0, stream>>>(Abf, Bbf, b_copy, hid, ws + OFF_SC);

  // joint softmax + outputs
  k_stats<<<1, 256, 0, stream>>>(ws + OFF_SG, ws + OFF_SC, ws + OFF_RED);
  k_probg<<<(kV + 255) / 256, 256, 0, stream>>>(ws + OFF_SG, ws + OFF_RED, out);
  k_scatter<<<8, 256, 0, stream>>>(ws + OFF_SC, ws + OFF_RED, idx, din, out, ws + OFF_WVEC);
  k_fw<<<dim3(4, 8), 256, 0, stream>>>(ws + OFF_WVEC, enc, out + kV + kH);
}
